// Transformer_9947144257631
// MI455X (gfx1250) — matrix operand order repaired and checked by reading, not compiled, not run on hardware
//
#include <hip/hip_runtime.h>
#include <hip/hip_bf16.h>

// ---------------------------------------------------------------------------
// Transformer forward (enc-dec, L=6, D=1024, H=16, B=4, S=512, FF=4096,
// V=32000) for MI455X (gfx1250, wave32, WMMA).
// All GEMMs run through v_wmma_f32_16x16x32_bf16 with fp32 accumulation;
// fp32 inputs are converted to bf16 during LDS staging.
// K-loop is software-pipelined: global prefetch of tile k+1 into registers
// overlaps the WMMA compute of tile k; ping-pong LDS buffers give exactly
// one barrier per K-step.
// ---------------------------------------------------------------------------

typedef __attribute__((ext_vector_type(16))) __bf16 v16bf;
typedef __attribute__((ext_vector_type(8)))  __bf16 v8bf;
typedef __attribute__((ext_vector_type(4)))  __bf16 v4bf;
typedef __attribute__((ext_vector_type(8)))  float  v8f;

#define TD   1024     // model dim
#define TFF  4096
#define THD  16       // heads
#define TDK  64
#define TB   4
#define TS   512      // seq len (both)
#define TL   6
#define TVT  32000
#define TEPS 0.001f
#define TSCALE 32.0f  // sqrt(1024)
#define TROWS (TB * TS)   // 2048

// ---------------------------------------------------------------------------
// WMMA GEMM:  C[m][n] = alpha * sum_k A[m][k] * B'[k][n] (+bias[n]) (ReLU?)
//   BTRANS=true : B is N x K row-major (weight W, C = A @ W^T)
//   BTRANS=false: B is K x N row-major (C = A @ B)
// BM=128, BK=32, 256 threads = 8 waves, wave grid 4(M) x 2(N).
// Strided-batched via gridDim.z with (b,h) = (z/BH, z%BH) decomposition.
// ---------------------------------------------------------------------------
template <int BN, bool BTRANS>
__global__ __launch_bounds__(256)
void gemm_bf16_wmma(const float* __restrict__ A, const float* __restrict__ B,
                    const float* __restrict__ bias, float* __restrict__ C,
                    int M, int N, int K, int lda, int ldb, int ldc,
                    long long sAb, long long sAh, long long sBb, long long sBh,
                    long long sCb, long long sCh, int BH,
                    float alpha, int relu) {
  constexpr int BM  = 128;
  constexpr int BK  = 32;
  constexpr int LDT = 40;            // padded bf16 row (80 B, 16B-aligned)
  constexpr int MT  = 2;             // 2 x 16 rows per wave
  constexpr int NT  = BN / 2 / 16;   // 16-col subtiles per wave
  constexpr int ANV = (BM * BK / 4) / 256;   // float4 prefetch regs for A
  constexpr int BNV = (BN * BK / 4) / 256;   // float4 prefetch regs for B

  __shared__ __align__(16) __bf16 As[2][BM * LDT];
  __shared__ __align__(16) __bf16 Bs[2][BN * LDT];

  const int z  = blockIdx.z;
  const int bb = z / BH;
  const int hh = z % BH;
  A += (long long)bb * sAb + (long long)hh * sAh;
  B += (long long)bb * sBb + (long long)hh * sBh;
  C += (long long)bb * sCb + (long long)hh * sCh;

  const int tid  = threadIdx.x;
  const int lane = tid & 31;
  const int wid  = tid >> 5;
  const int wm   = (wid & 3) * 32;         // wave M offset within block tile
  const int wn   = (wid >> 2) * (BN / 2);  // wave N offset
  const int rsel = lane & 15;
  const int hi   = lane >> 4;              // 0 / 1
  const int akh  = hi * 8;                 // A-frag k chunk base (ISA layout)
  const int bkh  = hi * 8;                 // B-frag k chunk base (ISA layout)

  const int m0 = blockIdx.y * BM;
  const int n0 = blockIdx.x * BN;

  v8f acc[MT][NT];
#pragma unroll
  for (int i = 0; i < MT; ++i)
#pragma unroll
    for (int j = 0; j < NT; ++j)
#pragma unroll
      for (int r = 0; r < 8; ++r) acc[i][j][r] = 0.0f;

  float4 ra[ANV];
  float4 rb[BNV];

  // ---- global -> register prefetch of one BK-wide tile ----
  auto gload = [&](int kb) {
#pragma unroll
    for (int t = 0; t < ANV; ++t) {
      int v   = tid + t * 256;
      int row = v >> 3;              // BK/4 = 8 float4 per row
      int kv  = (v & 7) << 2;
      ra[t] = *(const float4*)(A + (long long)(m0 + row) * lda + kb + kv);
    }
    if (BTRANS) {
#pragma unroll
      for (int t = 0; t < BNV; ++t) {
        int v   = tid + t * 256;
        int row = v >> 3;
        int kv  = (v & 7) << 2;
        rb[t] = *(const float4*)(B + (long long)(n0 + row) * ldb + kb + kv);
      }
    } else {
#pragma unroll
      for (int t = 0; t < BNV; ++t) {
        int v  = tid + t * 256;
        int kk = v / (BN / 4);
        int nv = (v % (BN / 4)) * 4;
        rb[t] = *(const float4*)(B + (long long)(kb + kk) * ldb + n0 + nv);
      }
    }
  };

  // ---- register -> LDS store with fp32 -> bf16 conversion ----
  auto lstore = [&](int buf) {
#pragma unroll
    for (int t = 0; t < ANV; ++t) {
      int v   = tid + t * 256;
      int row = v >> 3;
      int kv  = (v & 7) << 2;
      v4bf pk = { (__bf16)ra[t].x, (__bf16)ra[t].y, (__bf16)ra[t].z, (__bf16)ra[t].w };
      *(v4bf*)&As[buf][row * LDT + kv] = pk;
    }
    if (BTRANS) {
#pragma unroll
      for (int t = 0; t < BNV; ++t) {
        int v   = tid + t * 256;
        int row = v >> 3;
        int kv  = (v & 7) << 2;
        v4bf pk = { (__bf16)rb[t].x, (__bf16)rb[t].y, (__bf16)rb[t].z, (__bf16)rb[t].w };
        *(v4bf*)&Bs[buf][row * LDT + kv] = pk;
      }
    } else {
#pragma unroll
      for (int t = 0; t < BNV; ++t) {
        int v  = tid + t * 256;
        int kk = v / (BN / 4);
        int nv = (v % (BN / 4)) * 4;
        Bs[buf][(nv + 0) * LDT + kk] = (__bf16)rb[t].x;
        Bs[buf][(nv + 1) * LDT + kk] = (__bf16)rb[t].y;
        Bs[buf][(nv + 2) * LDT + kk] = (__bf16)rb[t].z;
        Bs[buf][(nv + 3) * LDT + kk] = (__bf16)rb[t].w;
      }
    }
  };

  // ---- pipeline prologue: tile 0 into buffer 0 ----
  gload(0);
  lstore(0);
  __syncthreads();

  const int nk = K / BK;
  for (int i = 0; i < nk; ++i) {
    const int cur = i & 1;
    const int nxt = cur ^ 1;

    // prefetch next tile while this tile computes
    if (i + 1 < nk) gload((i + 1) * BK);

    // ---- fragments (ISA 7.12.2 layouts) + WMMA ----
    v16bf af[MT], bf_[NT];
#pragma unroll
    for (int it = 0; it < MT; ++it) {
      const __bf16* p = &As[cur][(wm + it * 16 + rsel) * LDT];
      v8bf lo = *(const v8bf*)(p + akh);
      v8bf hc = *(const v8bf*)(p + akh + 16);
      af[it] = __builtin_shufflevector(lo, hc, 0, 1, 2, 3, 4, 5, 6, 7,
                                       8, 9, 10, 11, 12, 13, 14, 15);
    }
#pragma unroll
    for (int jt = 0; jt < NT; ++jt) {
      const __bf16* p = &Bs[cur][(wn + jt * 16 + rsel) * LDT];
      v8bf lo = *(const v8bf*)(p + bkh);
      v8bf hc = *(const v8bf*)(p + bkh + 16);
      bf_[jt] = __builtin_shufflevector(lo, hc, 0, 1, 2, 3, 4, 5, 6, 7,
                                        8, 9, 10, 11, 12, 13, 14, 15);
    }
#pragma unroll
    for (int it = 0; it < MT; ++it)
#pragma unroll
      for (int jt = 0; jt < NT; ++jt)
        acc[it][jt] = __builtin_amdgcn_wmma_f32_16x16x32_bf16(
            false, af[it], false, bf_[jt], (short)0, acc[it][jt], false, false);

    // publish next tile; single barrier per K-step (ping-pong is ordered by it)
    if (i + 1 < nk) {
      lstore(nxt);
      __syncthreads();
    }
  }

  // ---- epilogue: C fragment layout (VGPR r -> M = r + 8*hi) ----
#pragma unroll
  for (int it = 0; it < MT; ++it) {
#pragma unroll
    for (int jt = 0; jt < NT; ++jt) {
      const int mb = m0 + wm + it * 16 + hi * 8;
      const int nn = n0 + wn + jt * 16 + rsel;
      const float bv = bias ? bias[nn] : 0.0f;
#pragma unroll
      for (int r = 0; r < 8; ++r) {
        float vv = acc[it][jt][r] * alpha + bv;
        if (relu && vv < 0.0f) vv = 0.0f;
        C[(long long)(mb + r) * ldc + nn] = vv;
      }
    }
  }
}

// ---------------------------------------------------------------------------
// Embedding + sinusoidal positional encoding
// ---------------------------------------------------------------------------
__global__ __launch_bounds__(256)
void embed_pe_kernel(const float* __restrict__ emb, const int* __restrict__ toks,
                     float* __restrict__ out, int S) {
  const int rs  = blockIdx.x;       // b*S + s
  const int s   = rs % S;
  const int tok = toks[rs];
  const float* e = emb + (long long)tok * TD;
  float* o = out + (long long)rs * TD;
  const int nt = TD / 2;
  const float lti = __logf(10000.0f) / (float)(nt - 1);
  for (int d = threadIdx.x; d < TD; d += 256) {
    float p;
    if (d < nt) p = __sinf((float)s * __expf(-(float)d * lti));
    else        p = __cosf((float)s * __expf(-(float)(d - nt) * lti));
    o[d] = e[d] * TSCALE + p;
  }
}

// ---------------------------------------------------------------------------
// Masked softmax over S rows. mode: 0=enc self, 1=dec self (causal), 2=cross.
// Implements att = softmax(where(m, s, -inf)) * m.any(-1).
// ---------------------------------------------------------------------------
__global__ __launch_bounds__(256)
void softmax_mask_kernel(float* __restrict__ S, const int* __restrict__ qtok,
                         const int* __restrict__ ktok, int LQ, int LK, int H_,
                         int mode) {
  const int row = blockIdx.x;        // over B*H*LQ
  const int q   = row % LQ;
  const int zz  = row / LQ;
  const int b   = zz / H_;
  float* Sr = S + (long long)row * LK;
  const bool qv = qtok[(long long)b * LQ + q] != 0;

  __shared__ float red[256];
  const int tid = threadIdx.x;

  float mx = -3.0e38f;
  for (int k = tid; k < LK; k += 256) {
    bool kv = ktok[(long long)b * LK + k] != 0;
    bool m  = qv && kv && (mode != 1 || k <= q);
    if (m) mx = fmaxf(mx, Sr[k]);
  }
  red[tid] = mx;
  __syncthreads();
  for (int st = 128; st > 0; st >>= 1) {
    if (tid < st) red[tid] = fmaxf(red[tid], red[tid + st]);
    __syncthreads();
  }
  mx = red[0];
  const bool rowvalid = mx > -1.0e38f;
  __syncthreads();

  float sum = 0.0f;
  for (int k = tid; k < LK; k += 256) {
    bool kv = ktok[(long long)b * LK + k] != 0;
    bool m  = qv && kv && (mode != 1 || k <= q);
    float e = (m && rowvalid) ? __expf(Sr[k] - mx) : 0.0f;
    Sr[k] = e;
    sum += e;
  }
  red[tid] = sum;
  __syncthreads();
  for (int st = 128; st > 0; st >>= 1) {
    if (tid < st) red[tid] += red[tid + st];
    __syncthreads();
  }
  const float tot = red[0];
  const float inv = (rowvalid && tot > 0.0f) ? 1.0f / tot : 0.0f;
  __syncthreads();
  for (int k = tid; k < LK; k += 256) Sr[k] *= inv;
}

// ---------------------------------------------------------------------------
// out = LayerNorm(x + sub) * g + b  (sample std, /(n-1), EPS added to sigma)
// One block per row of D=1024, 256 threads, 4 elements/thread.
// ---------------------------------------------------------------------------
__global__ __launch_bounds__(256)
void add_layernorm_kernel(const float* __restrict__ X, const float* __restrict__ Sub,
                          const float* __restrict__ g, const float* __restrict__ bta,
                          float* __restrict__ Out) {
  const long long row = blockIdx.x;
  const float* x  = X + row * TD;
  const float* sb = Sub + row * TD;
  float* o = Out + row * TD;
  const int tid = threadIdx.x;

  __shared__ float red[256];
  float zv[4];
  float s = 0.0f;
#pragma unroll
  for (int i = 0; i < 4; ++i) {
    int d = tid + i * 256;
    float v = x[d] + sb[d];
    zv[i] = v;
    s += v;
  }
  red[tid] = s;
  __syncthreads();
  for (int st = 128; st > 0; st >>= 1) {
    if (tid < st) red[tid] += red[tid + st];
    __syncthreads();
  }
  const float mu = red[0] / (float)TD;
  __syncthreads();

  float ss = 0.0f;
#pragma unroll
  for (int i = 0; i < 4; ++i) {
    float c = zv[i] - mu;
    ss += c * c;
  }
  red[tid] = ss;
  __syncthreads();
  for (int st = 128; st > 0; st >>= 1) {
    if (tid < st) red[tid] += red[tid + st];
    __syncthreads();
  }
  const float sig = sqrtf(red[0] / (float)(TD - 1));
  const float inv = 1.0f / (sig + TEPS);
  __syncthreads();
#pragma unroll
  for (int i = 0; i < 4; ++i) {
    int d = tid + i * 256;
    o[d] = (zv[i] - mu) * inv * g[d] + bta[d];
  }
}

// ---------------------------------------------------------------------------
// Host-side helpers
// ---------------------------------------------------------------------------
static void run_gemm_nt(hipStream_t st, const float* A, const float* B,
                        const float* bias, float* C, int M, int N, int K,
                        int lda, int ldb, int ldc,
                        long long sAb, long long sAh, long long sBb, long long sBh,
                        long long sCb, long long sCh, int batch, int BH,
                        float alpha, int relu) {
  dim3 g(N / 128, M / 128, batch);
  gemm_bf16_wmma<128, true><<<g, 256, 0, st>>>(A, B, bias, C, M, N, K, lda, ldb,
                                               ldc, sAb, sAh, sBb, sBh, sCb, sCh,
                                               BH, alpha, relu);
}

static void run_gemm_nn64(hipStream_t st, const float* A, const float* B,
                          float* C, int M, int N, int K, int lda, int ldb,
                          int ldc, long long sAb, long long sAh, long long sBb,
                          long long sBh, long long sCb, long long sCh, int batch,
                          int BH, float alpha) {
  dim3 g(N / 64, M / 128, batch);
  gemm_bf16_wmma<64, false><<<g, 256, 0, st>>>(A, B, nullptr, C, M, N, K, lda,
                                               ldb, ldc, sAb, sAh, sBb, sBh, sCb,
                                               sCh, BH, alpha, 0);
}

// Full multi-head attention sublayer -> sub (2048 x 1024)
static void mha_block(hipStream_t st, const float* xq, const float* xkv,
                      const float* wq, const float* wk, const float* wv,
                      const float* wo, const int* qtok, const int* ktok,
                      int mode, float* qb, float* kb, float* vb, float* sc,
                      float* ctx, float* sub) {
  const int M = TROWS, Dd = TD, LQ = TS, LK = TS, BH = THD;
  // projections
  run_gemm_nt(st, xq,  wq, nullptr, qb, M, Dd, Dd, Dd, Dd, Dd, 0,0,0,0,0,0, 1, 1, 1.0f, 0);
  run_gemm_nt(st, xkv, wk, nullptr, kb, M, Dd, Dd, Dd, Dd, Dd, 0,0,0,0,0,0, 1, 1, 1.0f, 0);
  run_gemm_nt(st, xkv, wv, nullptr, vb, M, Dd, Dd, Dd, Dd, Dd, 0,0,0,0,0,0, 1, 1, 1.0f, 0);
  // S = Q K^T / sqrt(DK)   (batched over b,h)
  run_gemm_nt(st, qb, kb, nullptr, sc, LQ, LK, TDK, Dd, Dd, LK,
              (long long)LQ * Dd, TDK, (long long)LK * Dd, TDK,
              (long long)BH * LQ * LK, (long long)LQ * LK,
              TB * BH, BH, 0.125f, 0);
  softmax_mask_kernel<<<TB * BH * LQ, 256, 0, st>>>(sc, qtok, ktok, LQ, LK, BH, mode);
  // ctx = P V  (B is K x N here)
  run_gemm_nn64(st, sc, vb, ctx, LQ, TDK, LK, LK, Dd, Dd,
                (long long)BH * LQ * LK, (long long)LQ * LK,
                (long long)LK * Dd, TDK, (long long)LQ * Dd, TDK,
                TB * BH, BH, 1.0f);
  // sub = ctx @ Wo^T
  run_gemm_nt(st, ctx, wo, nullptr, sub, M, Dd, Dd, Dd, Dd, Dd, 0,0,0,0,0,0, 1, 1, 1.0f, 0);
}

extern "C" void kernel_launch(void* const* d_in, const int* in_sizes, int n_in,
                              void* d_out, int out_size, void* d_ws, size_t ws_size,
                              hipStream_t stream) {
  const float* emb_x    = (const float*)d_in[0];
  const float* emb_y    = (const float*)d_in[1];
  const float* enc_qkvo = (const float*)d_in[2];
  const float* enc_ln_s = (const float*)d_in[3];
  const float* enc_ln_b = (const float*)d_in[4];
  const float* enc_w1   = (const float*)d_in[5];
  const float* enc_b1   = (const float*)d_in[6];
  const float* enc_w2   = (const float*)d_in[7];
  const float* enc_b2   = (const float*)d_in[8];
  const float* dec_qkvo = (const float*)d_in[9];
  const float* dec_ln_s = (const float*)d_in[10];
  const float* dec_ln_b = (const float*)d_in[11];
  const float* dec_w1   = (const float*)d_in[12];
  const float* dec_b1   = (const float*)d_in[13];
  const float* dec_w2   = (const float*)d_in[14];
  const float* dec_b2   = (const float*)d_in[15];
  const int*   x_block  = (const int*)d_in[16];
  const int*   y_block  = (const int*)d_in[17];
  float* out = (float*)d_out;

  // --- workspace carve-out (~160 MB) ---
  char* w = (char*)d_ws;
  const size_t ACT = (size_t)TROWS * TD * sizeof(float);       // 8 MB
  float* e_enc = (float*)w; w += ACT;
  float* e_dec = (float*)w; w += ACT;
  float* qb    = (float*)w; w += ACT;
  float* kb    = (float*)w; w += ACT;
  float* vb    = (float*)w; w += ACT;
  float* ctx   = (float*)w; w += ACT;
  float* sub   = (float*)w; w += ACT;
  float* h1    = (float*)w; w += (size_t)TROWS * TFF * sizeof(float);   // 33.5 MB
  float* sc    = (float*)w; w += (size_t)TB * THD * TS * TS * sizeof(float); // 67 MB
  (void)w; (void)ws_size; (void)in_sizes; (void)n_in; (void)out_size;

  const long long DD = (long long)TD * TD;

  // --- embeddings ---
  embed_pe_kernel<<<TB * TS, 256, 0, stream>>>(emb_x, x_block, e_enc, TS);
  embed_pe_kernel<<<TB * TS, 256, 0, stream>>>(emb_y, y_block, e_dec, TS);

  // --- encoder ---
  for (int i = 0; i < TL; ++i) {
    const float* qk = enc_qkvo + (long long)i * 4 * DD;
    mha_block(stream, e_enc, e_enc, qk, qk + DD, qk + 2 * DD, qk + 3 * DD,
              x_block, x_block, /*mode=*/0, qb, kb, vb, sc, ctx, sub);
    add_layernorm_kernel<<<TROWS, 256, 0, stream>>>(
        e_enc, sub, enc_ln_s + (long long)(i * 2 + 0) * TD,
        enc_ln_b + (long long)(i * 2 + 0) * TD, e_enc);
    run_gemm_nt(stream, e_enc, enc_w1 + (long long)i * TFF * TD,
                enc_b1 + (long long)i * TFF, h1, TROWS, TFF, TD, TD, TD, TFF,
                0,0,0,0,0,0, 1, 1, 1.0f, /*relu=*/1);
    run_gemm_nt(stream, h1, enc_w2 + (long long)i * TD * TFF,
                enc_b2 + (long long)i * TD, sub, TROWS, TD, TFF, TFF, TFF, TD,
                0,0,0,0,0,0, 1, 1, 1.0f, 0);
    add_layernorm_kernel<<<TROWS, 256, 0, stream>>>(
        e_enc, sub, enc_ln_s + (long long)(i * 2 + 1) * TD,
        enc_ln_b + (long long)(i * 2 + 1) * TD, e_enc);
  }

  // --- decoder ---
  for (int i = 0; i < TL; ++i) {
    const float* qk = dec_qkvo + (long long)i * 8 * DD;
    // self attention (causal)
    mha_block(stream, e_dec, e_dec, qk, qk + DD, qk + 2 * DD, qk + 3 * DD,
              y_block, y_block, /*mode=*/1, qb, kb, vb, sc, ctx, sub);
    add_layernorm_kernel<<<TROWS, 256, 0, stream>>>(
        e_dec, sub, dec_ln_s + (long long)(i * 3 + 0) * TD,
        dec_ln_b + (long long)(i * 3 + 0) * TD, e_dec);
    // cross attention against encoder output
    mha_block(stream, e_dec, e_enc, qk + 4 * DD, qk + 5 * DD, qk + 6 * DD,
              qk + 7 * DD, y_block, x_block, /*mode=*/2, qb, kb, vb, sc, ctx, sub);
    add_layernorm_kernel<<<TROWS, 256, 0, stream>>>(
        e_dec, sub, dec_ln_s + (long long)(i * 3 + 1) * TD,
        dec_ln_b + (long long)(i * 3 + 1) * TD, e_dec);
    // FFN
    run_gemm_nt(stream, e_dec, dec_w1 + (long long)i * TFF * TD,
                dec_b1 + (long long)i * TFF, h1, TROWS, TFF, TD, TD, TD, TFF,
                0,0,0,0,0,0, 1, 1, 1.0f, /*relu=*/1);
    run_gemm_nt(stream, h1, dec_w2 + (long long)i * TD * TFF,
                dec_b2 + (long long)i * TD, sub, TROWS, TD, TFF, TFF, TFF, TD,
                0,0,0,0,0,0, 1, 1, 1.0f, 0);
    add_layernorm_kernel<<<TROWS, 256, 0, stream>>>(
        e_dec, sub, dec_ln_s + (long long)(i * 3 + 2) * TD,
        dec_ln_b + (long long)(i * 3 + 2) * TD, e_dec);
  }

  // --- logits = e_dec @ emb_y^T  -> (B, SY, VT) ---
  run_gemm_nt(stream, e_dec, emb_y, nullptr, out, TROWS, TVT, TD, TD, TD, TVT,
              0,0,0,0,0,0, 1, 1, 1.0f, 0);
}
